// RSSM_24163486008055
// MI455X (gfx1250) — matrix operand order repaired and checked by reading, not compiled, not run on hardware
//
#include <hip/hip_runtime.h>

// ---------------------------------------------------------------------------
// RSSM scan for MI455X (gfx1250). Weight-bandwidth-bound (B=16): convert all
// weights to bf16 once (140MB, L2-resident), run WMMA 16x16x32 bf16 GEMMs.
// Activation panel staged into LDS via TDM (tensor_load_to_lds).
// Each wave computes 4 N-tiles: one LDS A-fragment feeds 4 WMMAs.
// ---------------------------------------------------------------------------

typedef unsigned short u16;
typedef __attribute__((ext_vector_type(16))) __bf16 bf16x16;
typedef __attribute__((ext_vector_type(8)))  __bf16 bf16x8;
typedef __attribute__((ext_vector_type(8)))  float  f32x8;
typedef __attribute__((ext_vector_type(4)))  unsigned int u32x4;
typedef __attribute__((ext_vector_type(8)))  int i32x8;
typedef __attribute__((ext_vector_type(4)))  int i32x4;

#define BB 16
#define TT 64
#define EE 1024
#define AAA 32
#define DD 4096
#define HH 2048
#define SCC 1024
#define GGG 8
#define KDYN 6656            // D/G + 3H = 512 + 6144
#define OUTW (2*SCC + DD)    // 6144 floats per (b,t)
#define KCHUNK 1024          // A-panel K chunk staged in LDS (16*1024*2 = 32KB)

__device__ __forceinline__ u16 f2bf(float f) {
  unsigned u = __float_as_uint(f);
  unsigned r = (u + 0x7FFFu + ((u >> 16) & 1u)) >> 16;
  return (u16)r;
}
__device__ __forceinline__ float sigm(float x) { return 1.f / (1.f + __expf(-x)); }

// ---- Tensor Data Mover: 2D tile (rows x cols bf16) global -> LDS ----------
__device__ __forceinline__ void tdm_load_2d(unsigned lds_off, const u16* gp,
                                            int rows, int cols, long strideElems) {
  unsigned long long ga = (unsigned long long)(size_t)gp;
  u32x4 g0;
  g0[0] = 1u;                                   // count=1, user descriptor
  g0[1] = lds_off;                              // LDS byte address
  g0[2] = (unsigned)(ga & 0xFFFFFFFFull);       // global_addr[31:0]
  g0[3] = (unsigned)((ga >> 32) & 0x1FFFFFFull) | (2u << 30); // addr[56:32], type=2
  unsigned long long st = (unsigned long long)strideElems;
  i32x8 g1;
  g1[0] = (int)(1u << 16);                                  // data_size=1 (2B)
  g1[1] = (int)(((unsigned)cols & 0xFFFFu) << 16);          // tensor_dim0[15:0]
  g1[2] = (int)((((unsigned)cols >> 16) & 0xFFFFu) |
                (((unsigned)rows & 0xFFFFu) << 16));        // dim0[31:16], dim1[15:0]
  g1[3] = (int)(((unsigned)cols & 0xFFFFu) << 16);          // tile_dim0
  g1[4] = (int)((unsigned)rows & 0xFFFFu);                  // tile_dim1 (tile_dim2=0)
  g1[5] = (int)(st & 0xFFFFFFFFull);                        // dim0_stride[31:0]
  g1[6] = (int)((st >> 32) & 0xFFFFull);                    // dim0_stride[47:32]
  g1[7] = 0;
  i32x4 z4 = {0, 0, 0, 0};
#if __clang_major__ >= 23
  i32x8 z8 = {0, 0, 0, 0, 0, 0, 0, 0};
  __builtin_amdgcn_tensor_load_to_lds(g0, g1, z4, z4, z8, 0);
#else
  __builtin_amdgcn_tensor_load_to_lds(g0, g1, z4, z4, 0);
#endif
}

// ---------------------------------------------------------------------------
// GEMM: C[16, z*N + n] = A_bf16[16, K] @ WT_bf16[z][N, K] + bias[z*N + n]
// block = 128 threads = 4 waves; each wave owns 4 consecutive 16x16 tiles
// (64 columns) -> one A fragment per K-step feeds 4 WMMAs. grid.x = N/256.
// A panel staged in LDS per K-chunk via TDM; B read directly (L2-resident).
// ---------------------------------------------------------------------------
__global__ void __launch_bounds__(128) k_gemm(
    const u16* __restrict__ A, int lda, long aOff,
    const u16* __restrict__ WT, const float* __restrict__ bias,
    float* __restrict__ C, int ldc, long cOff, int K, int N) {
  __shared__ alignas(64) u16 sA[BB * KCHUNK];
  const int z = blockIdx.z;
  const u16* Az = A + (long)z * aOff;
  const u16* Wz = WT + (long)z * (long)K * (long)N;
  const int lane = threadIdx.x & 31;
  const int wave = threadIdx.x >> 5;
  const int row = lane & 15;   // A row / B column-in-tile / C column-in-tile
  const int hi  = lane >> 4;   // K-half select (and M+8 on output)
  const int col0 = (blockIdx.x * 16 + wave * 4) * 16 + row;  // tile j: col0+j*16
  f32x8 acc0 = {0.f, 0.f, 0.f, 0.f, 0.f, 0.f, 0.f, 0.f};
  f32x8 acc1 = acc0, acc2 = acc0, acc3 = acc0;
  const u16* bp0 = Wz + (long)(col0 +  0) * K + hi * 16;
  const u16* bp1 = Wz + (long)(col0 + 16) * K + hi * 16;
  const u16* bp2 = Wz + (long)(col0 + 32) * K + hi * 16;
  const u16* bp3 = Wz + (long)(col0 + 48) * K + hi * 16;

  for (int kc = 0; kc < K; kc += KCHUNK) {
    const int cw = (K - kc < KCHUNK) ? (K - kc) : KCHUNK;
    __syncthreads();                       // protect sA from previous chunk
    if (wave == 0) {
      tdm_load_2d((unsigned)(size_t)(&sA[0]), Az + kc, BB, cw, (long)lda);
      __builtin_amdgcn_s_wait_tensorcnt(0);
    }
    __syncthreads();
    if (kc + KCHUNK < K) {                 // pull next B chunk toward L2/L0
      __builtin_prefetch(bp0 + (kc + KCHUNK), 0, 1);
      __builtin_prefetch(bp1 + (kc + KCHUNK), 0, 1);
      __builtin_prefetch(bp2 + (kc + KCHUNK), 0, 1);
      __builtin_prefetch(bp3 + (kc + KCHUNK), 0, 1);
    }
#pragma unroll 2
    for (int kk = 0; kk < cw; kk += 32) {
      // A fragment in contiguous k order (B matches): lane<16 holds K[0:16); lane>=16 K[16:32)
      bf16x8 a0 = *(const bf16x8*)&sA[row * cw + kk + hi * 16];
      bf16x8 a1 = *(const bf16x8*)&sA[row * cw + kk + hi * 16 + 8];
      bf16x16 a = __builtin_shufflevector(a0, a1, 0, 1, 2, 3, 4, 5, 6, 7,
                                          8, 9, 10, 11, 12, 13, 14, 15);
      // B layout: lane = column, hi selects K half; WT is [N, K] -> contiguous
      bf16x16 b0 = *(const bf16x16*)(bp0 + kc + kk);
      bf16x16 b1 = *(const bf16x16*)(bp1 + kc + kk);
      bf16x16 b2 = *(const bf16x16*)(bp2 + kc + kk);
      bf16x16 b3 = *(const bf16x16*)(bp3 + kc + kk);
      acc0 = __builtin_amdgcn_wmma_f32_16x16x32_bf16(false, a, false, b0,
                                                     (short)0, acc0, false, false);
      acc1 = __builtin_amdgcn_wmma_f32_16x16x32_bf16(false, a, false, b1,
                                                     (short)0, acc1, false, false);
      acc2 = __builtin_amdgcn_wmma_f32_16x16x32_bf16(false, a, false, b2,
                                                     (short)0, acc2, false, false);
      acc3 = __builtin_amdgcn_wmma_f32_16x16x32_bf16(false, a, false, b3,
                                                     (short)0, acc3, false, false);
    }
  }
  const long zb = (long)z * N;
#pragma unroll
  for (int r = 0; r < 8; ++r) {
    int m = hi * 8 + r;                    // C/D layout: lane.hi -> rows 8..15
    long ro = cOff + (long)m * ldc + zb;
    C[ro + col0]      = acc0[r] + bias[zb + col0];
    C[ro + col0 + 16] = acc1[r] + bias[zb + col0 + 16];
    C[ro + col0 + 32] = acc2[r] + bias[zb + col0 + 32];
    C[ro + col0 + 48] = acc3[r] + bias[zb + col0 + 48];
  }
}

// ---- LayerNorm + SiLU over one row of N elements, write bf16 --------------
__global__ void __launch_bounds__(256) k_ln_silu(
    const float* __restrict__ X, int ldx, long xOff,
    const float* __restrict__ gam, const float* __restrict__ bet,
    u16* __restrict__ Y, int ldy, long yOff, int N) {
  __shared__ float sh[512];
  const int r = blockIdx.x;
  const float* xr = X + xOff + (long)r * ldx;
  float s = 0.f, s2 = 0.f;
  for (int i = threadIdx.x; i < N; i += 256) { float v = xr[i]; s += v; s2 += v * v; }
  sh[threadIdx.x] = s; sh[256 + threadIdx.x] = s2;
  __syncthreads();
  for (int st = 128; st > 0; st >>= 1) {
    if ((int)threadIdx.x < st) {
      sh[threadIdx.x] += sh[threadIdx.x + st];
      sh[256 + threadIdx.x] += sh[256 + threadIdx.x + st];
    }
    __syncthreads();
  }
  float mean = sh[0] / (float)N;
  float var  = sh[256] / (float)N - mean * mean;
  float inv  = rsqrtf(var + 1e-5f);
  u16* yr = Y + yOff + (long)r * ldy;
  for (int i = threadIdx.x; i < N; i += 256) {
    float v = (xr[i] - mean) * inv * gam[i] + bet[i];
    yr[i] = f2bf(v * sigm(v));
  }
}

// ---- weight fp32 -> bf16 with transpose: WT[z][n*K + k] = W[z][k*N + n] ---
__global__ void k_cvtT(const float* __restrict__ W, u16* __restrict__ T, int K, int N) {
  long z = blockIdx.z;
  const float* Wz = W + z * (long)K * N;
  u16* Tz = T + z * (long)K * N;
  long total = (long)K * N;
  for (long i = blockIdx.x * (long)blockDim.x + threadIdx.x; i < total;
       i += (long)gridDim.x * blockDim.x) {
    long n = i / K, k = i % K;
    Tz[i] = f2bf(Wz[k * (long)N + n]);
  }
}
__global__ void k_cvt(const float* __restrict__ X, u16* __restrict__ Y, long n) {
  for (long i = blockIdx.x * (long)blockDim.x + threadIdx.x; i < n;
       i += (long)gridDim.x * blockDim.x)
    Y[i] = f2bf(X[i]);
}

__global__ void k_init(float* dF, float* sF) {
  int i = blockIdx.x * blockDim.x + threadIdx.x;
  if (i < BB * DD) dF[i] = 0.f;
  if (i < BB * SCC) sF[i] = 0.f;
}

// ---- per-step mask + bf16 conversion of deter / stoch / action ------------
__global__ void k_prep(const float* __restrict__ action,
                       const unsigned char* __restrict__ reset, int t,
                       float* __restrict__ dF, float* __restrict__ sF,
                       u16* __restrict__ dB, u16* __restrict__ sB,
                       u16* __restrict__ aB) {
  int i = blockIdx.x * blockDim.x + threadIdx.x;
  const int totD = BB * DD, totS = BB * SCC, totA = BB * AAA;
  if (i < totD) {
    int b = i / DD;
    float m = reset[b * TT + t] ? 0.f : 1.f;
    float v = dF[i] * m; dF[i] = v; dB[i] = f2bf(v);
  } else if (i < totD + totS) {
    int j = i - totD; int b = j / SCC;
    float m = reset[b * TT + t] ? 0.f : 1.f;
    float v = sF[j] * m; sF[j] = v; sB[j] = f2bf(v);
  } else if (i < totD + totS + totA) {
    int j = i - totD - totS; int b = j / AAA; int c = j % AAA;
    float m = reset[b * TT + t] ? 0.f : 1.f;
    aB[j] = f2bf(action[((long)b * TT + t) * AAA + c] * m);
  }
}

// ---- dyn input: xg[g][16][6656] = [deter_g(512) | x(6144)] ----------------
__global__ void k_concat_xg(const u16* __restrict__ dB, const u16* __restrict__ xB,
                            u16* __restrict__ xg) {
  long i = blockIdx.x * (long)blockDim.x + threadIdx.x;
  const long per = (long)BB * KDYN;
  if (i >= GGG * per) return;
  int g = (int)(i / per);
  long rme = i % per;
  int r = (int)(rme / KDYN);
  int c = (int)(rme % KDYN);
  xg[i] = (c < 512) ? dB[r * DD + g * 512 + c] : xB[(long)r * (3 * HH) + (c - 512)];
}

// ---- posterior input: xin[16][5120] = [deter(4096) | embed_t(1024)] -------
__global__ void k_concat_post(const u16* __restrict__ dB, const u16* __restrict__ eB,
                              int t, u16* __restrict__ xin) {
  long i = blockIdx.x * (long)blockDim.x + threadIdx.x;
  if (i >= (long)BB * (DD + EE)) return;
  int r = (int)(i / (DD + EE)), c = (int)(i % (DD + EE));
  xin[i] = (c < DD) ? dB[r * DD + c] : eB[((long)r * TT + t) * EE + (c - DD)];
}

// ---- GRU update (flat-dim chunking, as reference) + write deter to out ----
__global__ void k_gru(const float* __restrict__ y, float* __restrict__ dF,
                      u16* __restrict__ dB, float* __restrict__ out, int t) {
  int i = blockIdx.x * blockDim.x + threadIdx.x;
  if (i >= BB * DD) return;
  int b = i / DD, j = i % DD;
  const float* yb = y + (long)b * 3 * DD;
  float rst  = sigm(yb[j]);
  float cand = tanhf(rst * yb[DD + j]);
  float upd  = sigm(yb[2 * DD + j] - 1.f);
  float dn = upd * cand + (1.f - upd) * dF[i];
  dF[i] = dn; dB[i] = f2bf(dn);
  out[((long)b * TT + t) * OUTW + 2 * SCC + j] = dn;
}

// ---- unimix softmax + categorical sample (hash RNG), onehot -> stoch ------
__global__ void k_sample(const float* __restrict__ out, int t, float* __restrict__ sF) {
  int i = blockIdx.x * blockDim.x + threadIdx.x;
  if (i >= BB * 32) return;
  int b = i / 32, s = i % 32;
  const float* lg = out + ((long)b * TT + t) * OUTW + s * 32;  // post logits
  float mx = -1e30f;
  for (int c = 0; c < 32; ++c) mx = fmaxf(mx, lg[c]);
  float e[32]; float sm = 0.f;
  for (int c = 0; c < 32; ++c) { e[c] = __expf(lg[c] - mx); sm += e[c]; }
  unsigned h = 0x9E3779B9u ^ (unsigned)(t * 73856093) ^ (unsigned)(b * 19349663)
             ^ (unsigned)(s * 83492791);
  h ^= h >> 16; h *= 0x7feb352du; h ^= h >> 15; h *= 0x846ca68bu; h ^= h >> 16;
  float u = (h >> 8) * (1.f / 16777216.f);
  float acc = 0.f; int pick = 31;
  for (int c = 0; c < 32; ++c) {
    float p = (e[c] / sm) * 0.99f + 0.01f / 32.f;
    acc += p;
    if (u < acc) { pick = c; break; }
  }
  float* dst = sF + b * SCC + s * 32;
  for (int c = 0; c < 32; ++c) dst[c] = (c == pick) ? 1.f : 0.f;
}

// ---------------------------------------------------------------------------
extern "C" void kernel_launch(void* const* d_in, const int* in_sizes, int n_in,
                              void* d_out, int out_size, void* d_ws, size_t ws_size,
                              hipStream_t stream) {
  (void)in_sizes; (void)n_in; (void)out_size; (void)ws_size;
  const float* embed  = (const float*)d_in[0];
  const float* action = (const float*)d_in[1];
  const unsigned char* reset = (const unsigned char*)d_in[2];
  const float *w_ind = (const float*)d_in[3],  *b_ind = (const float*)d_in[4],
              *g_ind = (const float*)d_in[5],  *v_ind = (const float*)d_in[6];
  const float *w_ins = (const float*)d_in[7],  *b_ins = (const float*)d_in[8],
              *g_ins = (const float*)d_in[9],  *v_ins = (const float*)d_in[10];
  const float *w_ina = (const float*)d_in[11], *b_ina = (const float*)d_in[12],
              *g_ina = (const float*)d_in[13], *v_ina = (const float*)d_in[14];
  const float *w_dyn = (const float*)d_in[15], *b_dyn = (const float*)d_in[16],
              *g_dyn = (const float*)d_in[17], *v_dyn = (const float*)d_in[18];
  const float *w_gru = (const float*)d_in[19], *b_gru = (const float*)d_in[20];
  const float *w_pr1 = (const float*)d_in[21], *b_pr1 = (const float*)d_in[22],
              *g_pr1 = (const float*)d_in[23], *v_pr1 = (const float*)d_in[24];
  const float *w_pr2 = (const float*)d_in[25], *b_pr2 = (const float*)d_in[26],
              *g_pr2 = (const float*)d_in[27], *v_pr2 = (const float*)d_in[28];
  const float *w_pr3 = (const float*)d_in[29], *b_pr3 = (const float*)d_in[30];
  const float *w_po1 = (const float*)d_in[31], *b_po1 = (const float*)d_in[32],
              *g_po1 = (const float*)d_in[33], *v_po1 = (const float*)d_in[34];
  const float *w_po2 = (const float*)d_in[35], *b_po2 = (const float*)d_in[36];
  float* out = (float*)d_out;

  // ---- workspace layout (bf16 region then fp32 region) --------------------
  u16* base = (u16*)d_ws;
  size_t o = 0;
  auto a16 = [&](size_t n) { u16* p = base + o; o += (n + 15) & ~(size_t)15; return p; };
  u16* wt_ind = a16((size_t)DD * HH);
  u16* wt_ins = a16((size_t)SCC * HH);
  u16* wt_ina = a16((size_t)AAA * HH);
  u16* wt_dyn = a16((size_t)GGG * KDYN * 512);
  u16* wt_gru = a16((size_t)GGG * 512 * 1536);
  u16* wt_pr1 = a16((size_t)DD * HH);
  u16* wt_pr2 = a16((size_t)HH * HH);
  u16* wt_pr3 = a16((size_t)HH * SCC);
  u16* wt_po1 = a16((size_t)(DD + EE) * HH);
  u16* wt_po2 = a16((size_t)HH * SCC);
  u16* eB  = a16((size_t)BB * TT * EE);
  u16* dB  = a16((size_t)BB * DD);
  u16* sB  = a16((size_t)BB * SCC);
  u16* aB  = a16((size_t)BB * AAA);
  u16* xB  = a16((size_t)BB * 3 * HH);
  u16* xgB = a16((size_t)GGG * BB * KDYN);
  u16* hB  = a16((size_t)BB * DD);
  u16* p1B = a16((size_t)BB * HH);
  u16* p2B = a16((size_t)BB * HH);
  u16* qB  = a16((size_t)BB * HH);
  u16* xinB = a16((size_t)BB * (DD + EE));
  float* fb = (float*)(base + ((o + 127) & ~(size_t)127));
  size_t fo = 0;
  auto aF = [&](size_t n) { float* p = fb + fo; fo += (n + 15) & ~(size_t)15; return p; };
  float* dF   = aF((size_t)BB * DD);
  float* sF   = aF((size_t)BB * SCC);
  float* xpre = aF((size_t)BB * 3 * HH);
  float* hpre = aF((size_t)BB * DD);
  float* ybuf = aF((size_t)BB * 3 * DD);
  float* p1p  = aF((size_t)BB * HH);
  float* p2p  = aF((size_t)BB * HH);
  float* qp   = aF((size_t)BB * HH);

  const dim3 cb(256);
  auto cvtT = [&](const float* W, u16* T, int K, int N, int batch) {
    k_cvtT<<<dim3(512, 1, batch), cb, 0, stream>>>(W, T, K, N);
  };
  cvtT(w_ind, wt_ind, DD, HH, 1);
  cvtT(w_ins, wt_ins, SCC, HH, 1);
  cvtT(w_ina, wt_ina, AAA, HH, 1);
  cvtT(w_dyn, wt_dyn, KDYN, 512, GGG);
  cvtT(w_gru, wt_gru, 512, 1536, GGG);
  cvtT(w_pr1, wt_pr1, DD, HH, 1);
  cvtT(w_pr2, wt_pr2, HH, HH, 1);
  cvtT(w_pr3, wt_pr3, HH, SCC, 1);
  cvtT(w_po1, wt_po1, DD + EE, HH, 1);
  cvtT(w_po2, wt_po2, HH, SCC, 1);
  k_cvt<<<dim3(1024), cb, 0, stream>>>(embed, eB, (long)BB * TT * EE);
  k_init<<<dim3((BB * DD + 255) / 256), cb, 0, stream>>>(dF, sF);

  auto gemm = [&](const u16* A, int lda, long aOff, const u16* WT, const float* bias,
                  float* C, int ldc, long cOff, int K, int N, int zb) {
    k_gemm<<<dim3(N / 256, 1, zb), dim3(128), 0, stream>>>(A, lda, aOff, WT, bias,
                                                           C, ldc, cOff, K, N);
  };
  auto ln = [&](const float* X, int ldx, long xOff, const float* g, const float* b,
                u16* Y, int ldy, long yOff, int N) {
    k_ln_silu<<<dim3(BB), dim3(256), 0, stream>>>(X, ldx, xOff, g, b, Y, ldy, yOff, N);
  };

  for (int t = 0; t < TT; ++t) {
    k_prep<<<dim3((BB * (DD + SCC + AAA) + 255) / 256), cb, 0, stream>>>(
        action, reset, t, dF, sF, dB, sB, aB);
    // input branches -> xpre[16, 6144]
    gemm(dB, DD, 0, wt_ind, b_ind, xpre, 3 * HH, 0,      DD,  HH, 1);
    gemm(sB, SCC, 0, wt_ins, b_ins, xpre, 3 * HH, HH,    SCC, HH, 1);
    gemm(aB, AAA, 0, wt_ina, b_ina, xpre, 3 * HH, 2 * HH, AAA, HH, 1);
    ln(xpre, 3 * HH, 0,      g_ind, v_ind, xB, 3 * HH, 0,      HH);
    ln(xpre, 3 * HH, HH,     g_ins, v_ins, xB, 3 * HH, HH,     HH);
    ln(xpre, 3 * HH, 2 * HH, g_ina, v_ina, xB, 3 * HH, 2 * HH, HH);
    // block-diagonal dyn
    k_concat_xg<<<dim3((GGG * BB * KDYN + 255) / 256), cb, 0, stream>>>(dB, xB, xgB);
    gemm(xgB, KDYN, (long)BB * KDYN, wt_dyn, b_dyn, hpre, DD, 0, KDYN, 512, GGG);
    ln(hpre, DD, 0, g_dyn, v_dyn, hB, DD, 0, DD);
    // block GRU
    gemm(hB, DD, 512, wt_gru, b_gru, ybuf, 3 * DD, 0, 512, 1536, GGG);
    k_gru<<<dim3((BB * DD + 255) / 256), cb, 0, stream>>>(ybuf, dF, dB, out, t);
    // prior head -> out[..., 1024:2048]
    gemm(dB, DD, 0, wt_pr1, b_pr1, p1p, HH, 0, DD, HH, 1);
    ln(p1p, HH, 0, g_pr1, v_pr1, p1B, HH, 0, HH);
    gemm(p1B, HH, 0, wt_pr2, b_pr2, p2p, HH, 0, HH, HH, 1);
    ln(p2p, HH, 0, g_pr2, v_pr2, p2B, HH, 0, HH);
    gemm(p2B, HH, 0, wt_pr3, b_pr3, out, TT * OUTW, (long)t * OUTW + SCC, HH, SCC, 1);
    // posterior head -> out[..., 0:1024]
    k_concat_post<<<dim3((BB * (DD + EE) + 255) / 256), cb, 0, stream>>>(dB, eB, t, xinB);
    gemm(xinB, DD + EE, 0, wt_po1, b_po1, qp, HH, 0, DD + EE, HH, 1);
    ln(qp, HH, 0, g_po1, v_po1, qB, HH, 0, HH);
    gemm(qB, HH, 0, wt_po2, b_po2, out, TT * OUTW, (long)t * OUTW, HH, SCC, 1);
    // sample next stoch (onehot forward value)
    k_sample<<<dim3((BB * 32 + 255) / 256), cb, 0, stream>>>(out, t, sF);
  }
}
